// GraphTransformer_25168508355097
// MI455X (gfx1250) — matrix operand order repaired and checked by reading, not compiled, not run on hardware
//
#include <hip/hip_runtime.h>
#include <hip/hip_bf16.h>

// ---------------------------------------------------------------------------
// GraphTransformer forward on MI455X (gfx1250), wave32 + WMMA f16->f32.
//   DIM=128, HEADS=8 (head_dim=128), DEPTH=5, MLP hidden=256, B=4, N=1024.
// Flash-style attention (no materialized 128MB score tensor), all GEMMs via
// v_wmma_f32_16x16x32_f16 with f32 accumulation. Fragment loads are 16-byte
// vector loads; GEMM waves are register-blocked 16x64 (A-frag reuse x4);
// attention V tiles are staged global->LDS with async LDS loads (ASYNCcnt).
// ---------------------------------------------------------------------------

typedef __attribute__((ext_vector_type(16))) _Float16 v16h;
typedef __attribute__((ext_vector_type(8)))  _Float16 h8;
typedef __attribute__((ext_vector_type(8)))  float    v8f;

#define NTOK   4096          // B*N tokens
#define SEQ    1024
#define NB     4
#define NH     8
#define HD     128
#define QKVLD  3072          // 3*NH*HD per token
#define KOFF   1024
#define VOFF   2048

static __device__ __forceinline__ v16h cat16(h8 lo, h8 hi) {
  return __builtin_shufflevector(lo, hi, 0,1,2,3,4,5,6,7,8,9,10,11,12,13,14,15);
}

// -------------------------------- f32 -> f16 -------------------------------
__global__ void k_f32_to_f16(const float* __restrict__ s, _Float16* __restrict__ d, long n) {
  long i = (long)blockIdx.x * blockDim.x + threadIdx.x;
  if (i < n) d[i] = (_Float16)s[i];
}

// f32 (rows x ks) -> f16 (rows x kd), zero-padding columns ks..kd-1
__global__ void k_pad_f32_to_f16(const float* __restrict__ s, _Float16* __restrict__ d,
                                 int rows, int ks, int kd) {
  int i = blockIdx.x * blockDim.x + threadIdx.x;
  if (i >= rows * kd) return;
  int r = i / kd, c = i - r * kd;
  d[i] = (c < ks) ? (_Float16)s[(long)r * ks + c] : (_Float16)0.f;
}

// ------------------------------- LayerNorm ---------------------------------
__global__ __launch_bounds__(32) void k_ln(const float* __restrict__ x,
    const float* __restrict__ w, const float* __restrict__ b,
    float* __restrict__ y32, _Float16* __restrict__ y16) {
  const int t = blockIdx.x;
  const int lane = threadIdx.x;
  float4 v = ((const float4*)(x + (long)t * 128))[lane];
  float s  = v.x + v.y + v.z + v.w;
  float ss = v.x*v.x + v.y*v.y + v.z*v.z + v.w*v.w;
  #pragma unroll
  for (int m = 1; m < 32; m <<= 1) { s += __shfl_xor(s, m); ss += __shfl_xor(ss, m); }
  float mean = s * (1.0f/128.0f);
  float var  = ss * (1.0f/128.0f) - mean*mean;
  float inv  = rsqrtf(var + 1e-5f);
  float xv[4] = {v.x, v.y, v.z, v.w};
  #pragma unroll
  for (int i = 0; i < 4; ++i) {
    int d = lane*4 + i;
    float yv = (xv[i] - mean) * inv * w[d] + b[d];
    if (y32) y32[(long)t*128 + d] = yv;
    if (y16) y16[(long)t*128 + d] = (_Float16)yv;
  }
}

// ------------------------------ gamma = exp(h@gW.T+gb) ---------------------
__global__ __launch_bounds__(32) void k_gamma(const float* __restrict__ hln,
    const float* __restrict__ gw, const float* __restrict__ gb,
    float* __restrict__ gam) {
  const int t = blockIdx.x;
  const int lane = threadIdx.x;
  float4 v  = ((const float4*)(hln + (long)t*128))[lane];
  float4 w0 = ((const float4*)gw)[lane];
  float4 w1 = ((const float4*)(gw + 128))[lane];
  float s0 = v.x*w0.x + v.y*w0.y + v.z*w0.z + v.w*w0.w;
  float s1 = v.x*w1.x + v.y*w1.y + v.z*w1.z + v.w*w1.w;
  #pragma unroll
  for (int m = 1; m < 32; m <<= 1) { s0 += __shfl_xor(s0, m); s1 += __shfl_xor(s1, m); }
  if (lane == 0) {
    gam[(long)t*2 + 0] = expf(s0 + gb[0]);
    gam[(long)t*2 + 1] = expf(s1 + gb[1]);
  }
}

// --------------------------- generic WMMA GEMM -----------------------------
// C = act( A(MxK,f16) @ W(NxK,f16)^T + bias + resid ), write f32 and/or f16.
// K % 32 == 0. 4 waves/block; each wave computes a 16x(16*NT) strip and
// reuses one A fragment across NT WMMAs (register blocking along N).
template<int NT>
__global__ __launch_bounds__(128) void k_gemm(const _Float16* __restrict__ A,
    const _Float16* __restrict__ W, const float* __restrict__ bias,
    const float* __restrict__ resid, float* __restrict__ C32,
    _Float16* __restrict__ C16, int M, int N, int K, int act) {
  const int lane  = threadIdx.x & 31;
  const int wid   = threadIdx.x >> 5;
  const int nl    = lane & 15;
  const int half  = lane >> 4;
  const int nbase = (blockIdx.x * 4 + wid) * (16 * NT);
  const int mbase = blockIdx.y * 16;
  v8f c[NT];
  #pragma unroll
  for (int t = 0; t < NT; ++t) c[t] = (v8f){0.f,0.f,0.f,0.f,0.f,0.f,0.f,0.f};
  const _Float16* Ap = A + (long)(mbase + nl) * K + 16*half;  // A rows (lane&15)
  const _Float16* Wp = W + (long)(nbase + nl) * K + 16*half;  // B cols (lane&15)
  for (int k0 = 0; k0 < K; k0 += 32) {
    // A frag: lanes hold K = 16*half+[0..15]
    v16h a = cat16(*(const h8*)(Ap + k0), *(const h8*)(Ap + k0 + 8));
    #pragma unroll
    for (int t = 0; t < NT; ++t) {
      const _Float16* Wpt = Wp + (long)t * 16 * K;
      v16h bm = cat16(*(const h8*)(Wpt + k0), *(const h8*)(Wpt + k0 + 8));
      c[t] = __builtin_amdgcn_wmma_f32_16x16x32_f16(false, a, false, bm, (short)0,
                                                    c[t], false, false);
    }
  }
  #pragma unroll
  for (int t = 0; t < NT; ++t) {
    const int n = nbase + t*16 + nl;
    const float bv = bias ? bias[n] : 0.f;
    #pragma unroll
    for (int r = 0; r < 8; ++r) {
      int m = mbase + r + 8*half;        // C layout: VGPR r -> row r (+8 hi half)
      float v = c[t][r] + bv;
      if (resid) v += resid[(long)m*N + n];
      if (act == 1)      v = fmaxf(v, 0.f);
      else if (act == 2) v = 0.5f * v * (1.f + erff(v * 0.70710678f));
      if (C32) C32[(long)m*N + n] = v;
      if (C16) C16[(long)m*N + n] = (_Float16)v;
    }
  }
}

// ----------------------- flash attention (one wave / 16 queries) -----------
// softmax( g0 * QK^T/sqrt(128) + g1 * adj ) @ V  -> o16 (tok, h*128+d)
__global__ __launch_bounds__(32) void k_attn(const _Float16* __restrict__ qkv,
    const float* __restrict__ adj, const float* __restrict__ gam,
    _Float16* __restrict__ o16) {
  __shared__ _Float16 P[16*32];     // probabilities (transposed staging)
  __shared__ _Float16 Vt[32*128];   // current V tile, row-major
  const int lane  = threadIdx.x & 31;
  const int nl    = lane & 15;
  const int half  = lane >> 4;
  const int qbase = blockIdx.x * 16;
  const int h     = blockIdx.y;
  const int b     = blockIdx.z;
  const long tokb = (long)b * SEQ;
  const float scl = 0.08838834764831845f;   // 128^-0.5

  // Q fragments: 4 chunks of K=32 over head_dim 128 (16B vector loads)
  v16h qa[4];
  const _Float16* Qp = qkv + (tokb + qbase + nl) * QKVLD + h * HD + 16*half;
  #pragma unroll
  for (int cfr = 0; cfr < 4; ++cfr)
    qa[cfr] = cat16(*(const h8*)(Qp + cfr*32), *(const h8*)(Qp + cfr*32 + 8));

  float mrow[8], lrow[8], g0v[8], g1v[8];
  v8f o[8];
  #pragma unroll
  for (int r = 0; r < 8; ++r) {
    mrow[r] = -3.0e38f; lrow[r] = 0.f;
    int q = qbase + r + 8*half;
    g0v[r] = gam[(tokb + q)*2 + 0];
    g1v[r] = gam[(tokb + q)*2 + 1];
  }
  #pragma unroll
  for (int nbk = 0; nbk < 8; ++nbk) o[nbk] = (v8f){0.f,0.f,0.f,0.f,0.f,0.f,0.f,0.f};

  const unsigned ldsV = (unsigned)(size_t)(&Vt[lane * 128]);  // per-lane LDS row addr

  for (int kb = 0; kb < SEQ; kb += 32) {
    // stage V tile (32 keys x 128 f16) global->LDS with async loads (ASYNCcnt):
    // each lane DMA-copies its row in 16B chunks, no VGPR round-trip.
    {
      unsigned long long ga =
          (unsigned long long)(size_t)(qkv + (tokb + kb + lane) * QKVLD + VOFF + h * HD);
      unsigned la = ldsV;
      #pragma unroll
      for (int ch = 0; ch < 16; ++ch) {
        asm volatile("global_load_async_to_lds_b128 %0, %1, off"
                     :: "v"(la), "v"(ga) : "memory");
        la += 16; ga += 16;
      }
    }
    // scores for 32 keys: two 16x16 WMMA accumulators
    v8f s0 = {0.f,0.f,0.f,0.f,0.f,0.f,0.f,0.f};
    v8f s1 = {0.f,0.f,0.f,0.f,0.f,0.f,0.f,0.f};
    const _Float16* Kp0 = qkv + (tokb + kb + nl) * QKVLD + KOFF + h * HD + 16*half;
    const _Float16* Kp1 = Kp0 + 16 * QKVLD;
    __builtin_prefetch(Kp0 + 32 * QKVLD, 0, 3);   // next K tile
    #pragma unroll
    for (int cfr = 0; cfr < 4; ++cfr) {
      v16h b0 = cat16(*(const h8*)(Kp0 + cfr*32), *(const h8*)(Kp0 + cfr*32 + 8));
      v16h b1 = cat16(*(const h8*)(Kp1 + cfr*32), *(const h8*)(Kp1 + cfr*32 + 8));
      s0 = __builtin_amdgcn_wmma_f32_16x16x32_f16(false, qa[cfr], false, b0, (short)0, s0, false, false);
      s1 = __builtin_amdgcn_wmma_f32_16x16x32_f16(false, qa[cfr], false, b1, (short)0, s1, false, false);
    }
    // online softmax over this 16x32 block (row = r + 8*half, cols across lanes)
    #pragma unroll
    for (int r = 0; r < 8; ++r) {
      int q = qbase + r + 8*half;
      const float* adjrow = adj + ((long)b*SEQ + q)*SEQ + kb;
      if (r == 0) __builtin_prefetch(adjrow + 32, 0, 3);   // next adj block
      float x0 = g0v[r]*scl*s0[r] + g1v[r]*adjrow[nl];
      float x1 = g0v[r]*scl*s1[r] + g1v[r]*adjrow[16 + nl];
      float rm = fmaxf(x0, x1);
      rm = fmaxf(rm, __shfl_xor(rm, 1)); rm = fmaxf(rm, __shfl_xor(rm, 2));
      rm = fmaxf(rm, __shfl_xor(rm, 4)); rm = fmaxf(rm, __shfl_xor(rm, 8));
      float mn = fmaxf(mrow[r], rm);
      float alpha = expf(mrow[r] - mn);
      float p0 = expf(x0 - mn), p1 = expf(x1 - mn);
      float ps = p0 + p1;
      ps += __shfl_xor(ps, 1); ps += __shfl_xor(ps, 2);
      ps += __shfl_xor(ps, 4); ps += __shfl_xor(ps, 8);
      lrow[r] = lrow[r]*alpha + ps;
      mrow[r] = mn;
      int row = r + 8*half;
      P[row*32 + nl]      = (_Float16)p0;
      P[row*32 + 16 + nl] = (_Float16)p1;
      #pragma unroll
      for (int nbk = 0; nbk < 8; ++nbk) o[nbk][r] *= alpha;
    }
    // async V-stage + P transpose stores must be visible before LDS reads
    asm volatile("s_wait_asynccnt 0x0\n\ts_wait_dscnt 0x0" ::: "memory");
    // re-load P as A-matrix fragment (transpose via LDS), two 16B ds loads
    v16h pa = cat16(*(const h8*)(&P[nl*32 + 16*half]),
                    *(const h8*)(&P[nl*32 + 16*half + 8]));
    // O += P(16x32) @ V(32x128), 8 column tiles; V frags from LDS
    #pragma unroll
    for (int nbk = 0; nbk < 8; ++nbk) {
      v16h vb;
      #pragma unroll
      for (int j = 0; j < 8; ++j) {
        int p = 16*half + 2*j;
        vb[2*j]   = Vt[p*128       + nbk*16 + nl];
        vb[2*j+1] = Vt[(p+1)*128   + nbk*16 + nl];
      }
      o[nbk] = __builtin_amdgcn_wmma_f32_16x16x32_f16(false, pa, false, vb, (short)0, o[nbk], false, false);
    }
  }
  // normalize + store (B, N, H*HD)
  #pragma unroll
  for (int r = 0; r < 8; ++r) {
    float inv = 1.0f / lrow[r];
    int q = qbase + r + 8*half;
    _Float16* op = o16 + (tokb + q)*(NH*HD) + h*HD;
    #pragma unroll
    for (int nbk = 0; nbk < 8; ++nbk) op[nbk*16 + nl] = (_Float16)(o[nbk][r] * inv);
  }
}

// ------------------------- final 128 -> 1 projection -----------------------
__global__ __launch_bounds__(32) void k_dot_out(const float* __restrict__ xin,
    const float* __restrict__ w, const float* __restrict__ bsc, float* __restrict__ out) {
  const int t = blockIdx.x;
  const int lane = threadIdx.x;
  float4 v  = ((const float4*)(xin + (long)t*128))[lane];
  float4 wv = ((const float4*)w)[lane];
  float s = v.x*wv.x + v.y*wv.y + v.z*wv.z + v.w*wv.w;
  #pragma unroll
  for (int m = 1; m < 32; m <<= 1) s += __shfl_xor(s, m);
  if (lane == 0) out[t] = s + bsc[0];
}

// ---------------------------------------------------------------------------
static inline void* wsAlloc(char*& p, size_t bytes) {
  void* r = (void*)p;
  p += (bytes + 255) & ~(size_t)255;
  return r;
}

extern "C" void kernel_launch(void* const* d_in, const int* in_sizes, int n_in,
                              void* d_out, int out_size, void* d_ws, size_t ws_size,
                              hipStream_t stream) {
  (void)in_sizes; (void)n_in; (void)out_size; (void)ws_size;
  // --- inputs (setup_inputs dict order, params flattened in insertion order)
  const float* x       = (const float*)d_in[0];
  const float* adj     = (const float*)d_in[1];
  const float* n1w     = (const float*)d_in[2];
  const float* n1b     = (const float*)d_in[3];
  const float* qkv_w   = (const float*)d_in[4];
  const float* proj_w  = (const float*)d_in[5];
  const float* proj_b  = (const float*)d_in[6];
  const float* g_w     = (const float*)d_in[7];
  const float* g_b     = (const float*)d_in[8];
  const float* n2w     = (const float*)d_in[9];
  const float* n2b     = (const float*)d_in[10];
  const float* w1      = (const float*)d_in[11];
  const float* b1      = (const float*)d_in[12];
  const float* w2      = (const float*)d_in[13];
  const float* b2      = (const float*)d_in[14];
  const float* emb_w1  = (const float*)d_in[15];
  const float* emb_b1  = (const float*)d_in[16];
  const float* emb_w2  = (const float*)d_in[17];
  const float* emb_b2  = (const float*)d_in[18];
  const float* hlnw    = (const float*)d_in[19];
  const float* hlnb    = (const float*)d_in[20];
  const float* head_w  = (const float*)d_in[21];
  const float* head_b  = (const float*)d_in[22];
  const float* p1_w    = (const float*)d_in[23];
  const float* p1_b    = (const float*)d_in[24];
  const float* p2_w    = (const float*)d_in[25];
  const float* p2_b    = (const float*)d_in[26];
  const float* p3_w    = (const float*)d_in[27];
  const float* p3_b    = (const float*)d_in[28];
  const float* plnw    = (const float*)d_in[29];
  const float* plnb    = (const float*)d_in[30];
  const float* p4_w    = (const float*)d_in[31];
  const float* p4_b    = (const float*)d_in[32];
  float* out = (float*)d_out;

  // --- workspace layout
  char* wp = (char*)d_ws;
  _Float16* qkvw16 = (_Float16*)wsAlloc(wp, (size_t)5*3072*128*2);
  _Float16* projw16= (_Float16*)wsAlloc(wp, (size_t)5*128*1024*2);
  _Float16* w1_16  = (_Float16*)wsAlloc(wp, (size_t)5*256*128*2);
  _Float16* w2_16  = (_Float16*)wsAlloc(wp, (size_t)5*128*256*2);
  _Float16* headw16= (_Float16*)wsAlloc(wp, (size_t)128*128*2);
  _Float16* p1w16  = (_Float16*)wsAlloc(wp, (size_t)128*128*2);
  _Float16* p2w16  = (_Float16*)wsAlloc(wp, (size_t)128*128*2);
  _Float16* p3w16  = (_Float16*)wsAlloc(wp, (size_t)128*128*2);
  _Float16* embw1p = (_Float16*)wsAlloc(wp, (size_t)256*64*2);   // padded K=36->64
  _Float16* embw216= (_Float16*)wsAlloc(wp, (size_t)128*256*2);
  _Float16* x16p   = (_Float16*)wsAlloc(wp, (size_t)NTOK*64*2);  // padded K=36->64
  float*    h      = (float*)   wsAlloc(wp, (size_t)NTOK*128*4);
  float*    hln32  = (float*)   wsAlloc(wp, (size_t)NTOK*128*4);
  _Float16* hln16  = (_Float16*)wsAlloc(wp, (size_t)NTOK*128*2);
  float*    gam    = (float*)   wsAlloc(wp, (size_t)NTOK*2*4);
  _Float16* qkv16  = (_Float16*)wsAlloc(wp, (size_t)NTOK*3072*2);
  _Float16* o16    = (_Float16*)wsAlloc(wp, (size_t)NTOK*1024*2);
  float*    hattn  = (float*)   wsAlloc(wp, (size_t)NTOK*128*4);
  float*    h2_32  = (float*)   wsAlloc(wp, (size_t)NTOK*128*4);
  _Float16* h2_16  = (_Float16*)wsAlloc(wp, (size_t)NTOK*128*2);
  _Float16* mlp16  = (_Float16*)wsAlloc(wp, (size_t)NTOK*256*2);
  _Float16* t16a   = (_Float16*)wsAlloc(wp, (size_t)NTOK*128*2);
  _Float16* t16b   = (_Float16*)wsAlloc(wp, (size_t)NTOK*128*2);
  float*    t32a   = (float*)   wsAlloc(wp, (size_t)NTOK*128*4);
  float*    t32b   = (float*)   wsAlloc(wp, (size_t)NTOK*128*4);

  auto cvt = [&](const float* s, _Float16* d, long n) {
    k_f32_to_f16<<<dim3((unsigned)((n + 255) / 256)), dim3(256), 0, stream>>>(s, d, n);
  };
  // --- one-time (per launch) f32->f16 weight/input conversion
  cvt(qkv_w,  qkvw16, (long)5*3072*128);
  cvt(proj_w, projw16,(long)5*128*1024);
  cvt(w1,     w1_16,  (long)5*256*128);
  cvt(w2,     w2_16,  (long)5*128*256);
  cvt(head_w, headw16,128*128);
  cvt(p1_w,   p1w16,  128*128);
  cvt(p2_w,   p2w16,  128*128);
  cvt(p3_w,   p3w16,  128*128);
  cvt(emb_w2, embw216,128*256);
  k_pad_f32_to_f16<<<dim3((256*64 + 255)/256), dim3(256), 0, stream>>>(emb_w1, embw1p, 256, 36, 64);
  k_pad_f32_to_f16<<<dim3((NTOK*64 + 255)/256), dim3(256), 0, stream>>>(x, x16p, NTOK, 36, 64);

  // N % 256 == 0 -> NT=4 (wave strip 16x64); N == 128 -> NT=2 (strip 16x32)
  auto gemm = [&](const _Float16* A, const _Float16* W, const float* bias,
                  const float* resid, float* C32, _Float16* C16,
                  int M, int N, int K, int act) {
    if (N % 256 == 0)
      k_gemm<4><<<dim3(N/256, M/16), dim3(128), 0, stream>>>(A, W, bias, resid, C32, C16, M, N, K, act);
    else
      k_gemm<2><<<dim3(N/128, M/16), dim3(128), 0, stream>>>(A, W, bias, resid, C32, C16, M, N, K, act);
  };

  // --- embedding: relu(x @ w1^T + b1) @ w2^T + b2 -> h   (K padded to 64)
  gemm(x16p,  embw1p,  emb_b1, nullptr, nullptr, mlp16, NTOK, 256, 64,  1);
  gemm(mlp16, embw216, emb_b2, nullptr, h,       nullptr, NTOK, 128, 256, 0);

  // --- transformer blocks
  for (int dep = 0; dep < 5; ++dep) {
    k_ln<<<dim3(NTOK), dim3(32), 0, stream>>>(h, n1w + dep*128, n1b + dep*128, hln32, hln16);
    k_gamma<<<dim3(NTOK), dim3(32), 0, stream>>>(hln32, g_w + dep*256, g_b + dep*2, gam);
    gemm(hln16, qkvw16 + (size_t)dep*3072*128, nullptr, nullptr, nullptr, qkv16,
         NTOK, 3072, 128, 0);
    k_attn<<<dim3(SEQ/16, NH, NB), dim3(32), 0, stream>>>(qkv16, adj, gam, o16);
    gemm(o16, projw16 + (size_t)dep*128*1024, proj_b + dep*128, hln32, hattn, nullptr,
         NTOK, 128, 1024, 0);
    k_ln<<<dim3(NTOK), dim3(32), 0, stream>>>(hattn, n2w + dep*128, n2b + dep*128, h2_32, h2_16);
    gemm(h2_16, w1_16 + (size_t)dep*256*128, b1 + dep*256, nullptr, nullptr, mlp16,
         NTOK, 256, 128, 2);
    gemm(mlp16, w2_16 + (size_t)dep*128*256, b2 + dep*128, h2_32, h, nullptr,
         NTOK, 128, 256, 0);
  }

  // --- head
  k_ln<<<dim3(NTOK), dim3(32), 0, stream>>>(h, hlnw, hlnb, nullptr, hln16);
  gemm(hln16, headw16, head_b, nullptr, nullptr, t16a, NTOK, 128, 128, 0);
  gemm(t16a,  p1w16,   p1_b,   nullptr, nullptr, t16b, NTOK, 128, 128, 2);
  gemm(t16b,  p2w16,   p2_b,   nullptr, nullptr, t16a, NTOK, 128, 128, 2);
  gemm(t16a,  p3w16,   p3_b,   nullptr, t32a,    nullptr, NTOK, 128, 128, 2);
  k_ln<<<dim3(NTOK), dim3(32), 0, stream>>>(t32a, plnw, plnb, t32b, nullptr);
  k_dot_out<<<dim3(NTOK), dim3(32), 0, stream>>>(t32b, p4_w, p4_b, out);
}
